// Encoder_49615462203897
// MI455X (gfx1250) — compile-verified, matrix operand order checked by reading, not run on hardware
//
#include <hip/hip_runtime.h>
#include <hip/hip_bf16.h>
#include <stdint.h>

// Problem constants (match reference)
#define NN 50000
#define CC 32
#define TT 12
#define DD 128
#define EE 1600000
#define KK 3
#define GATES 4
#define NOUT (GATES*DD)      // 512 = 4 gates * 128
#define KX (KK*CC)           // 96
#define KH (KK*DD)           // 384
#define MPAD 50048           // 64*782, pad M so 64-row wave strips need no guards
#define LN_EPS 1e-5f

typedef __attribute__((ext_vector_type(16))) __bf16 v16bf;
typedef __attribute__((ext_vector_type(8)))  float  v8f;

// ---------------------------------------------------------------- utilities
__global__ void zero_f32(float* __restrict__ p, long long n) {
    long long i = (long long)blockIdx.x * blockDim.x + threadIdx.x;
    if (i < n) p[i] = 0.0f;
}

// deg[src] += 1
__global__ void deg_kernel(const int* __restrict__ ei, float* __restrict__ deg, int E) {
    int e = blockIdx.x * blockDim.x + threadIdx.x;
    if (e < E) atomicAdd(&deg[ei[e]], 1.0f);
}

// in-place deg -> dis = deg>0 ? rsqrt(max(deg,1e-12)) : 0
__global__ void dis_kernel(float* __restrict__ deg, int N) {
    int n = blockIdx.x * blockDim.x + threadIdx.x;
    if (n < N) {
        float d = deg[n];
        deg[n] = (d > 0.0f) ? rsqrtf(fmaxf(d, 1e-12f)) : 0.0f;
    }
}

// w[e] = -dis[src]*dis[dst]
__global__ void edgew_kernel(const int* __restrict__ ei, const float* __restrict__ dis,
                             float* __restrict__ w, int E) {
    int e = blockIdx.x * blockDim.x + threadIdx.x;
    if (e < E) w[e] = -dis[ei[e]] * dis[ei[EE + e]];
}

// SpMM: out[dst] += w[e] * v[src]  (F floats per row, float4 per thread).
// chunks threads share one edge -> row gather and atomic scatter are coalesced;
// operand rows (<=25.6MB) are L2-resident on MI455X (192MB L2).
__global__ void spmm_kernel(const int* __restrict__ ei, const float* __restrict__ w,
                            const float* __restrict__ v, float* __restrict__ out,
                            int E, int F) {
    int chunks = F >> 2;
    long long tid = (long long)blockIdx.x * blockDim.x + threadIdx.x;
    if (tid >= (long long)E * chunks) return;
    int e  = (int)(tid / chunks);
    int f4 = (int)(tid % chunks);
    int s = ei[e], d = ei[EE + e];
    float we = w[e];
    float4 vv = reinterpret_cast<const float4*>(v + (size_t)s * F)[f4];
    float* o = out + (size_t)d * F + (size_t)f4 * 4;
    atomicAdd(o + 0, we * vv.x);
    atomicAdd(o + 1, we * vv.y);
    atomicAdd(o + 2, we * vv.z);
    atomicAdd(o + 3, we * vv.w);
}

// Tx0[n,c] = x[n,c,t]   (x is [N,C,T])
__global__ void gather_xt(const float* __restrict__ x, float* __restrict__ Tx0,
                          int N, int C, int T, int t) {
    int i = blockIdx.x * blockDim.x + threadIdx.x;
    if (i >= N * C) return;
    Tx0[i] = x[(size_t)i * T + t];
}

// T2 = 2*T2 - T0
__global__ void cheb_combine(float* __restrict__ T2, const float* __restrict__ T0, long long n) {
    long long i = (long long)blockIdx.x * blockDim.x + threadIdx.x;
    if (i < n) T2[i] = 2.0f * T2[i] - T0[i];
}

// A[n, kc*F + f] = bf16( Tkc[n*F + f] )  -> bf16 A-stack [MPAD rows, 3F] (first N rows written)
__global__ void packA_bf16(const float* __restrict__ T0, const float* __restrict__ T1,
                           const float* __restrict__ T2, __hip_bfloat16* __restrict__ A,
                           int N, int F) {
    long long i = (long long)blockIdx.x * blockDim.x + threadIdx.x;
    long long total = (long long)N * 3 * F;
    if (i >= total) return;
    int r  = (int)(i % (3 * F));
    long long n = i / (3 * F);
    int kc = r / F, f = r % F;
    const float* src = (kc == 0) ? T0 : ((kc == 1) ? T1 : T2);
    A[i] = __float2bfloat16(src[n * F + f]);
}

// Repack W [4, Kcheb, Cin, 128] f32 -> WMMA-B-native packed bf16:
// per 32-deep K-step ks, per column n, per lane-half h, 8 dwords laid out exactly
// as the B operand VGPRs (j<4: pair k=ks*32+h*8+2j ; j>=4: pair k=ks*32+16+h*8+2(j-4)),
// so a lane fetches its whole operand with two contiguous b128 loads.
__global__ void repack_W(const float* __restrict__ W, __hip_bfloat16* __restrict__ Bp,
                         int Kcheb, int Cin) {
    int idx = blockIdx.x * blockDim.x + threadIdx.x;
    int total = GATES * Kcheb * Cin * DD;
    if (idx >= total) return;
    int d = idx % DD;
    int t = idx / DD;
    int c = t % Cin; t /= Cin;
    int kc = t % Kcheb;
    int g  = t / Kcheb;
    int k = kc * Cin + c;          // GEMM K index
    int n = g * DD + d;            // GEMM N index
    int ks = k >> 5;
    int kr = k & 31;
    int h, j;
    if (kr < 16) { h = kr >> 3;        j =      (kr & 7) >> 1; }
    else         { int q = kr - 16; h = q >> 3; j = 4 + ((q & 7) >> 1); }
    int p = k & 1;
    Bp[((((size_t)ks * NOUT + n) * 2 + h) * 8 + j) * 2 + p] = __float2bfloat16(W[idx]);
}

// ------------------------------------------------------------- WMMA GEMM
// C[MPAD, NOUT] (+)= A[MPAD, KD] @ Bp, bf16 operands, f32 accum.
// One wave32 computes a 64x16 strip (4 stacked 16x16 tiles). Per 32-deep K step:
// B operand = 2 x b128 (reused by 4 v_wmma), each A operand = 2 x b128.
// Base pointers hoisted -> all K-loop loads are constant immediate offsets.
// Register double-buffering of both A and B operands; waves_per_eu(2) grants the
// RA enough VGPRs (~acc 32 + operands 80 + ptrs) to keep the pipeline live, so
// step ks+1's 10 loads issue before step ks's 4 WMMAs and latency hides.
template <int KD>
__global__ __launch_bounds__(256)
__attribute__((amdgpu_waves_per_eu(2)))
void wmma_gemm_bf16(const __hip_bfloat16* __restrict__ A,
                    const __hip_bfloat16* __restrict__ Bp,
                    float* __restrict__ C,
                    int beta) {
    const int lane = threadIdx.x & 31;
    const int wave = threadIdx.x >> 5;
    const int gw   = blockIdx.x * 8 + wave;
    const int ntn  = NOUT >> 4;                 // 32 n-tiles
    const int mt   = gw / ntn;
    const int nt   = gw % ntn;
    const int mbase = mt * 64;
    if (mbase >= MPAD) return;

    const int l16  = lane & 15;
    const int half = lane >> 4;
    const int ncol = nt * 16 + l16;

    union ACC { v8f v; float f[8]; };
    ACC acc[4];
    if (beta) {
#pragma unroll
        for (int s = 0; s < 4; ++s)
#pragma unroll
            for (int r = 0; r < 8; ++r)
                acc[s].f[r] = C[(size_t)(mbase + s * 16 + r + half * 8) * NOUT + ncol];
    } else {
#pragma unroll
        for (int s = 0; s < 4; ++s)
#pragma unroll
            for (int r = 0; r < 8; ++r) acc[s].f[r] = 0.0f;
    }

    const uint4* A4 = reinterpret_cast<const uint4*>(A);   // 1 uint4 = 8 bf16
    const uint4* B4 = reinterpret_cast<const uint4*>(Bp);

    // Hoisted base pointers: all K-loop loads become constant immediate offsets.
    const uint4* aP[4];
#pragma unroll
    for (int s = 0; s < 4; ++s)
        aP[s] = A4 + (((size_t)(mbase + s * 16 + l16) * KD + half * 8) >> 3);
    const uint4* bP = B4 + ((size_t)(ncol * 2 + half) * 2);

    union AB { v16bf v; uint4 q[2]; };
    constexpr int NS = KD / 32;
    constexpr int BSTEP = NOUT * 4;   // uint4 stride of Bp per 32-deep K step

    AB bbuf[2];
    AB abuf[2][4];

    auto load_step = [&](int ks, AB& b, AB a[4]) {
        b.q[0] = bP[ks * BSTEP];
        b.q[1] = bP[ks * BSTEP + 1];
#pragma unroll
        for (int s = 0; s < 4; ++s) {
            a[s].q[0] = aP[s][ks * 4];        // K pairs kbase .. kbase+7
            a[s].q[1] = aP[s][ks * 4 + 2];    // K pairs kbase+16 .. kbase+23
        }
    };

    load_step(0, bbuf[0], abuf[0]);
#pragma unroll
    for (int ks = 0; ks < NS; ++ks) {
        const int cur = ks & 1;
        const int nxt = cur ^ 1;
        if (ks + 1 < NS) load_step(ks + 1, bbuf[nxt], abuf[nxt]);
#pragma unroll
        for (int s = 0; s < 4; ++s)
            acc[s].v = __builtin_amdgcn_wmma_f32_16x16x32_bf16(
                false, abuf[cur][s].v, false, bbuf[cur].v, (short)0, acc[s].v, false, false);
    }

#pragma unroll
    for (int s = 0; s < 4; ++s)
#pragma unroll
        for (int r = 0; r < 8; ++r)
            C[(size_t)(mbase + s * 16 + r + half * 8) * NOUT + ncol] = acc[s].f[r];
}

// ------------------------------------------- fused gates + peephole + tanh + LN
__global__ __launch_bounds__(128)
void gate_ln(const float* __restrict__ G,      // [MPAD, 512] = gates i,f,c,o (Gx+Gh)
             const float* __restrict__ bx, const float* __restrict__ bh,
             const float* __restrict__ wc, const float* __restrict__ bg,
             const float* __restrict__ gamma, const float* __restrict__ lbeta,
             float* __restrict__ h, float* __restrict__ c,
             float* __restrict__ out, int N, int t) {
    int n = blockIdx.x, d = threadIdx.x;
    const float* g = G + (size_t)n * NOUT;
    float cv = c[(size_t)n * DD + d];

    float gi = g[d]        + bx[d]        + bh[d]        + wc[d]        * cv + bg[d];
    float gf = g[128 + d]  + bx[128 + d]  + bh[128 + d]  + wc[128 + d]  * cv + bg[128 + d];
    float gc = g[256 + d]  + bx[256 + d]  + bh[256 + d]                      + bg[256 + d];
    float i_ = 1.0f / (1.0f + __expf(-gi));
    float f_ = 1.0f / (1.0f + __expf(-gf));
    float cn = f_ * cv + i_ * tanhf(gc);
    float go = g[384 + d] + bx[384 + d] + bh[384 + d] + wc[256 + d] * cn + bg[384 + d];
    float o_ = 1.0f / (1.0f + __expf(-go));
    float hn = o_ * tanhf(cn);

    h[(size_t)n * DD + d] = hn;
    c[(size_t)n * DD + d] = cn;

    float y = tanhf(hn);
    __shared__ float s1[DD], s2[DD];
    s1[d] = y; s2[d] = y * y;
    __syncthreads();
    for (int s = DD / 2; s > 0; s >>= 1) {
        if (d < s) { s1[d] += s1[d + s]; s2[d] += s2[d + s]; }
        __syncthreads();
    }
    float mu  = s1[0] * (1.0f / DD);
    float var = s2[0] * (1.0f / DD) - mu * mu;
    float r   = rsqrtf(var + LN_EPS);
    out[((size_t)t * N + n) * DD + d] = (y - mu) * r * gamma[d] + lbeta[d];
}

// ---------------------------------------------------------------- driver
static inline int cdiv_ll(long long a, int b) { return (int)((a + b - 1) / b); }

extern "C" void kernel_launch(void* const* d_in, const int* in_sizes, int n_in,
                              void* d_out, int out_size, void* d_ws, size_t ws_size,
                              hipStream_t stream) {
    const float* x     = (const float*)d_in[0];   // [N,C,T]
    const int*   ei    = (const int*)  d_in[1];   // [2,E]
    const float* Wx    = (const float*)d_in[2];   // [4,K,C,D]
    const float* bx    = (const float*)d_in[3];   // [4,D]
    const float* Wh    = (const float*)d_in[4];   // [4,K,D,D]
    const float* bh    = (const float*)d_in[5];   // [4,D]
    const float* wc    = (const float*)d_in[6];   // [3,D]
    const float* bg    = (const float*)d_in[7];   // [4,D]
    const float* gamma = (const float*)d_in[8];   // [D]
    const float* lbeta = (const float*)d_in[9];   // [D]
    float* out = (float*)d_out;                   // [T,N,D]

    // carve workspace
    char* ws = (char*)d_ws;
    auto carve = [&](size_t bytes) -> char* {
        char* p = ws;
        ws += (bytes + 255) & ~(size_t)255;
        return p;
    };
    float* deg = (float*)carve((size_t)NN * 4);            // deg -> dis (in place)
    float* w   = (float*)carve((size_t)EE * 4);
    float* Tx0 = (float*)carve((size_t)NN * CC * 4);
    float* Tx1 = (float*)carve((size_t)NN * CC * 4);
    float* Tx2 = (float*)carve((size_t)NN * CC * 4);
    float* Th1 = (float*)carve((size_t)NN * DD * 4);
    float* Th2 = (float*)carve((size_t)NN * DD * 4);
    float* h   = (float*)carve((size_t)NN * DD * 4);
    float* c   = (float*)carve((size_t)NN * DD * 4);
    float* G   = (float*)carve((size_t)MPAD * NOUT * 4);
    __hip_bfloat16* Ax  = (__hip_bfloat16*)carve((size_t)MPAD * KX * 2);
    __hip_bfloat16* Ah  = (__hip_bfloat16*)carve((size_t)MPAD * KH * 2);
    __hip_bfloat16* WxP = (__hip_bfloat16*)carve((size_t)KX * NOUT * 2);
    __hip_bfloat16* WhP = (__hip_bfloat16*)carve((size_t)KH * NOUT * 2);

    const int B = 256;

    // edge normalization
    zero_f32<<<cdiv_ll(NN, B), B, 0, stream>>>(deg, NN);
    deg_kernel<<<cdiv_ll(EE, B), B, 0, stream>>>(ei, deg, EE);
    dis_kernel<<<cdiv_ll(NN, B), B, 0, stream>>>(deg, NN);
    edgew_kernel<<<cdiv_ll(EE, B), B, 0, stream>>>(ei, deg, w, EE);

    // repack weights for WMMA B operand
    repack_W<<<cdiv_ll((long long)GATES * KK * CC * DD, B), B, 0, stream>>>(Wx, WxP, KK, CC);
    repack_W<<<cdiv_ll((long long)GATES * KK * DD * DD, B), B, 0, stream>>>(Wh, WhP, KK, DD);

    // init state
    zero_f32<<<cdiv_ll((long long)NN * DD, B), B, 0, stream>>>(h, (long long)NN * DD);
    zero_f32<<<cdiv_ll((long long)NN * DD, B), B, 0, stream>>>(c, (long long)NN * DD);

    const long long nc = (long long)NN * CC;
    const long long nd = (long long)NN * DD;
    const int spmm_x_blocks = cdiv_ll((long long)EE * (CC / 4), B);
    const int spmm_h_blocks = cdiv_ll((long long)EE * (DD / 4), B);
    const int gemm_blocks   = ((MPAD / 64) * (NOUT / 16)) / 8;   // 8 waves/block

    for (int t = 0; t < TT; ++t) {
        // --- Chebyshev basis for x_t ---
        gather_xt<<<cdiv_ll(nc, B), B, 0, stream>>>(x, Tx0, NN, CC, TT, t);
        zero_f32<<<cdiv_ll(nc, B), B, 0, stream>>>(Tx1, nc);
        spmm_kernel<<<spmm_x_blocks, B, 0, stream>>>(ei, w, Tx0, Tx1, EE, CC);
        zero_f32<<<cdiv_ll(nc, B), B, 0, stream>>>(Tx2, nc);
        spmm_kernel<<<spmm_x_blocks, B, 0, stream>>>(ei, w, Tx1, Tx2, EE, CC);
        cheb_combine<<<cdiv_ll(nc, B), B, 0, stream>>>(Tx2, Tx0, nc);
        packA_bf16<<<cdiv_ll(nc * 3, B), B, 0, stream>>>(Tx0, Tx1, Tx2, Ax, NN, CC);
        wmma_gemm_bf16<KX><<<gemm_blocks, B, 0, stream>>>(Ax, WxP, G, /*beta=*/0);

        // --- Chebyshev basis for h ---
        zero_f32<<<cdiv_ll(nd, B), B, 0, stream>>>(Th1, nd);
        spmm_kernel<<<spmm_h_blocks, B, 0, stream>>>(ei, w, h, Th1, EE, DD);
        zero_f32<<<cdiv_ll(nd, B), B, 0, stream>>>(Th2, nd);
        spmm_kernel<<<spmm_h_blocks, B, 0, stream>>>(ei, w, Th1, Th2, EE, DD);
        cheb_combine<<<cdiv_ll(nd, B), B, 0, stream>>>(Th2, h, nd);
        packA_bf16<<<cdiv_ll(nd * 3, B), B, 0, stream>>>(h, Th1, Th2, Ah, NN, DD);
        wmma_gemm_bf16<KH><<<gemm_blocks, B, 0, stream>>>(Ah, WhP, G, /*beta=*/1);

        // --- fused LSTM gates + peephole + tanh + LayerNorm ---
        gate_ln<<<NN, DD, 0, stream>>>(G, bx, bh, wc, bg, gamma, lbeta, h, c, out, NN, t);
    }
}
